// CtmpGIN_53008486367430
// MI455X (gfx1250) — matrix operand order repaired and checked by reading, not compiled, not run on hardware
//
#include <hip/hip_runtime.h>

// ---------------- problem constants (match reference) ----------------
#define BATCH   1024
#define NCOLS   128
#define NN      64
#define EMB     64
#define HDIM    256
#define CARD    100
#define NEDGE   1000000
#define NNODE   (BATCH * 2 * NN)   // 131072
#define LN_EPS  1e-5f

typedef __attribute__((ext_vector_type(16))) __bf16 bf16x16;
typedef __attribute__((ext_vector_type(8)))  __bf16 bf16x8;
typedef __attribute__((ext_vector_type(8)))  float  f32x8;

// ------------------------------------------------------------------
// Weight repack: row-major f32 (K x Nc) -> per-lane WMMA B fragments.
// Fragment (kt, nt): lane l holds col n = nt*16 + (l&15),
// 16 contiguous K values k = kt*32 + (l>>4)*16 + j  (j = 0..15),
// the same contiguous k order load_a_frag uses for A.
// ------------------------------------------------------------------
__global__ void repack_w_kernel(const float* __restrict__ W,
                                __bf16* __restrict__ rep, int K, int Nc) {
    int o = blockIdx.x * 256 + threadIdx.x;
    if (o >= K * Nc) return;
    int NT   = Nc >> 4;
    int frag = o >> 9;          // /512 (32 lanes * 16 vals)
    int lane = (o >> 4) & 31;
    int j    = o & 15;
    int kt = frag / NT, nt = frag % NT;
    int k  = kt * 32 + ((lane >> 4) << 4) + j;
    int n  = nt * 16 + (lane & 15);
    rep[o] = (__bf16)W[(size_t)k * Nc + n];
}

// ------------------------------------------------------------------
// Embedding gather: node n -> emb_table[x[b, n%128] + (n%128)*CARD]
// ------------------------------------------------------------------
__global__ void embed_kernel(const int* __restrict__ x,
                             const float* __restrict__ emb,
                             float* __restrict__ h) {
    int idx = blockIdx.x * 256 + threadIdx.x;       // NNODE*16 float4 chunks
    int n  = idx >> 4;
    int ch = idx & 15;
    int b   = n >> 7;
    int col = n & 127;
    int row = x[b * 128 + col] + col * CARD;
    float4 v = *(const float4*)(emb + (size_t)row * EMB + ch * 4);
    *(float4*)(h + (size_t)n * EMB + ch * 4) = v;
}

// ------------------------------------------------------------------
// z = (1 + eps[l]) * h        (scatter adds on top afterwards)
// ------------------------------------------------------------------
__global__ void init_z_kernel(const float* __restrict__ h,
                              const float* __restrict__ eps, int l,
                              float* __restrict__ z, int total) {
    int i = blockIdx.x * 256 + threadIdx.x;
    if (i < total) z[i] = (1.0f + eps[l]) * h[i];
}

// ------------------------------------------------------------------
// Edge scatter: z[dst] += h[src], float4 gather + 4x f32 atomics.
// Node table (<=128MB) is L2-resident on MI455X (192MB L2).
// ------------------------------------------------------------------
__global__ void scatter_kernel(const float* __restrict__ h,
                               const int* __restrict__ src,
                               const int* __restrict__ dst,
                               float* __restrict__ z,
                               int H, int shift, unsigned total) {
    unsigned i = blockIdx.x * 256u + threadIdx.x;
    if (i >= total) return;
    int e  = (int)(i >> shift);
    int ch = (int)(i & ((1u << shift) - 1u));
    int s = src[e], d = dst[e];
    float4 v = *(const float4*)(h + (size_t)s * H + ch * 4);
    float* p = z + (size_t)d * H + ch * 4;
    atomicAdd(p + 0, v.x);
    atomicAdd(p + 1, v.y);
    atomicAdd(p + 2, v.z);
    atomicAdd(p + 3, v.w);
}

__global__ void f32_to_bf16_kernel(const float* __restrict__ in,
                                   __bf16* __restrict__ out, int total) {
    int i = blockIdx.x * 256 + threadIdx.x;
    if (i < total) out[i] = (__bf16)in[i];
}

// ------------------------------------------------------------------
// A-fragment loader: 16x32 bf16 A layout.
// lane l: row m = l&15; koff = (l>>4)*16;
// halves 0-7  -> K = kt*32 + koff + j        (VGPR 0-3)
// halves 8-15 -> K = kt*32 + koff + 8 + j    (VGPR 4-7)
// ------------------------------------------------------------------
__device__ __forceinline__ bf16x16 load_a_frag(const __bf16* __restrict__ A,
                                               int row, int Hin, int kt, int lane) {
    int koff = (lane >> 4) << 4;
    const __bf16* p = A + (size_t)row * Hin + kt * 32 + koff;
    union { bf16x16 full; bf16x8 half[2]; } u;
    u.half[0] = *(const bf16x8*)(p);
    u.half[1] = *(const bf16x8*)(p + 8);
    return u.full;
}

// ------------------------------------------------------------------
// GEMM1 fused: Out = relu(LN(A @ W + bias) * gamma + beta) -> bf16
// Block: 256 threads (8 waves), 16 rows x 256 cols of output.
// Each wave owns 2 column tiles; K loop over Hin/32 WMMA steps.
// ------------------------------------------------------------------
__global__ __launch_bounds__(256) void gemm_ln_relu_kernel(
    const __bf16* __restrict__ A, const __bf16* __restrict__ Brep,
    const float* __restrict__ bias, const float* __restrict__ gamma,
    const float* __restrict__ beta, __bf16* __restrict__ Out, int Hin) {
    __shared__ float tile[16][HDIM + 4];
    __shared__ float psum[16][17];
    __shared__ float psqr[16][17];
    __shared__ float s_mean[16], s_inv[16];

    const int lane = threadIdx.x & 31;
    const int wave = threadIdx.x >> 5;
    const int rowbase = blockIdx.x * 16;
    const int NT = HDIM >> 4;               // 16 column tiles
    const int nt0 = wave * 2, nt1 = wave * 2 + 1;
    const int KT = Hin >> 5;

    f32x8 acc0 = {}, acc1 = {};
    for (int kt = 0; kt < KT; ++kt) {
        bf16x16 a  = load_a_frag(A, rowbase + (lane & 15), Hin, kt, lane);
        bf16x16 b0 = *(const bf16x16*)(Brep + ((size_t)(kt * NT + nt0) * 32 + lane) * 16);
        bf16x16 b1 = *(const bf16x16*)(Brep + ((size_t)(kt * NT + nt1) * 32 + lane) * 16);
        acc0 = __builtin_amdgcn_wmma_f32_16x16x32_bf16(false, a, false, b0,
                                                       (short)0, acc0, false, false);
        acc1 = __builtin_amdgcn_wmma_f32_16x16x32_bf16(false, a, false, b1,
                                                       (short)0, acc1, false, false);
    }
    // bias + stash tiles to LDS (C layout: VGPR v -> M = v + (lane>>4)*8, N = lane&15)
    {
        int n0 = nt0 * 16 + (lane & 15);
        int n1 = nt1 * 16 + (lane & 15);
        float bb0 = bias[n0], bb1 = bias[n1];
        int mh = (lane >> 4) << 3;
#pragma unroll
        for (int v = 0; v < 8; ++v) {
            tile[mh + v][n0] = acc0[v] + bb0;
            tile[mh + v][n1] = acc1[v] + bb1;
        }
    }
    __syncthreads();
    // LayerNorm over each 256-wide row: partial sums (16 threads / row)
    {
        int r  = threadIdx.x >> 4;
        int cb = (threadIdx.x & 15) << 4;
        float s = 0.f, s2 = 0.f;
#pragma unroll
        for (int j = 0; j < 16; ++j) {
            float xv = tile[r][cb + j];
            s += xv; s2 += xv * xv;
        }
        psum[r][threadIdx.x & 15] = s;
        psqr[r][threadIdx.x & 15] = s2;
    }
    __syncthreads();
    if (threadIdx.x < 16) {
        float s = 0.f, s2 = 0.f;
#pragma unroll
        for (int j = 0; j < 16; ++j) { s += psum[threadIdx.x][j]; s2 += psqr[threadIdx.x][j]; }
        float mean = s * (1.0f / 256.0f);
        float var  = s2 * (1.0f / 256.0f) - mean * mean;
        s_mean[threadIdx.x] = mean;
        s_inv[threadIdx.x]  = rsqrtf(var + LN_EPS);
    }
    __syncthreads();
    {
        int r  = threadIdx.x >> 4;
        int cb = (threadIdx.x & 15) << 4;
        float mean = s_mean[r], inv = s_inv[r];
        __bf16* outp = Out + (size_t)(rowbase + r) * HDIM + cb;
#pragma unroll
        for (int j = 0; j < 16; ++j) {
            int c = cb + j;
            float v = (tile[r][c] - mean) * inv * gamma[c] + beta[c];
            v = v > 0.f ? v : 0.f;
            outp[j] = (__bf16)v;
        }
    }
}

// ------------------------------------------------------------------
// GEMM2: Out = A @ W + bias  (f32 out), same tiling, Hin = 256.
// ------------------------------------------------------------------
__global__ __launch_bounds__(256) void gemm_bias_kernel(
    const __bf16* __restrict__ A, const __bf16* __restrict__ Brep,
    const float* __restrict__ bias, float* __restrict__ Out) {
    const int lane = threadIdx.x & 31;
    const int wave = threadIdx.x >> 5;
    const int rowbase = blockIdx.x * 16;
    const int NT = HDIM >> 4;
    const int nt0 = wave * 2, nt1 = wave * 2 + 1;

    f32x8 acc0 = {}, acc1 = {};
    for (int kt = 0; kt < (HDIM >> 5); ++kt) {
        bf16x16 a  = load_a_frag(A, rowbase + (lane & 15), HDIM, kt, lane);
        bf16x16 b0 = *(const bf16x16*)(Brep + ((size_t)(kt * NT + nt0) * 32 + lane) * 16);
        bf16x16 b1 = *(const bf16x16*)(Brep + ((size_t)(kt * NT + nt1) * 32 + lane) * 16);
        acc0 = __builtin_amdgcn_wmma_f32_16x16x32_bf16(false, a, false, b0,
                                                       (short)0, acc0, false, false);
        acc1 = __builtin_amdgcn_wmma_f32_16x16x32_bf16(false, a, false, b1,
                                                       (short)0, acc1, false, false);
    }
    int n0 = nt0 * 16 + (lane & 15);
    int n1 = nt1 * 16 + (lane & 15);
    float bb0 = bias[n0], bb1 = bias[n1];
    int mh = (lane >> 4) << 3;
#pragma unroll
    for (int v = 0; v < 8; ++v) {
        Out[(size_t)(rowbase + mh + v) * HDIM + n0] = acc0[v] + bb0;
        Out[(size_t)(rowbase + mh + v) * HDIM + n1] = acc1[v] + bb1;
    }
}

// ------------------------------------------------------------------
// Final: out[g, c] = sum over 64 nodes of h[(g*64+n), c]
// ------------------------------------------------------------------
__global__ void graph_sum_kernel(const float* __restrict__ h,
                                 float* __restrict__ out) {
    int idx = blockIdx.x * 256 + threadIdx.x;   // 2048*256
    int g = idx >> 8, c = idx & 255;
    const float* p = h + (size_t)g * 64 * HDIM + c;
    float s = 0.f;
#pragma unroll 8
    for (int n = 0; n < 64; ++n) s += p[(size_t)n * HDIM];
    out[idx] = s;
}

static inline int cdiv(long long a, long long b) { return (int)((a + b - 1) / b); }

extern "C" void kernel_launch(void* const* d_in, const int* in_sizes, int n_in,
                              void* d_out, int out_size, void* d_ws, size_t ws_size,
                              hipStream_t stream) {
    const int*   x    = (const int*)d_in[0];
    const int*   edge = (const int*)d_in[1];   // [2, E]: src then dst
    const float* emb  = (const float*)d_in[3]; // d_in[2] (los) unused by reference
    const float* W0a  = (const float*)d_in[4];
    const float* b0a  = (const float*)d_in[5];
    const float* g0   = (const float*)d_in[6];
    const float* be0  = (const float*)d_in[7];
    const float* W0b  = (const float*)d_in[8];
    const float* b0b  = (const float*)d_in[9];
    const float* Wha  = (const float*)d_in[10];
    const float* bha  = (const float*)d_in[11];
    const float* gh   = (const float*)d_in[12];
    const float* beh  = (const float*)d_in[13];
    const float* Whb  = (const float*)d_in[14];
    const float* bhb  = (const float*)d_in[15];
    const float* eps  = (const float*)d_in[16];

    char* ws = (char*)d_ws;
    float*  h    = (float*)ws;  ws += (size_t)NNODE * HDIM * 4;
    float*  z    = (float*)ws;  ws += (size_t)NNODE * HDIM * 4;
    __bf16* abf  = (__bf16*)ws; ws += (size_t)NNODE * HDIM * 2;
    __bf16* tbf  = (__bf16*)ws; ws += (size_t)NNODE * HDIM * 2;
    __bf16* w0aR = (__bf16*)ws; ws += (size_t)EMB  * HDIM * 2;
    __bf16* w0bR = (__bf16*)ws; ws += (size_t)HDIM * HDIM * 2;
    __bf16* whaR = (__bf16*)ws; ws += (size_t)HDIM * HDIM * 2;
    __bf16* whbR = (__bf16*)ws; ws += (size_t)HDIM * HDIM * 2;

    const int* srcI = edge;
    const int* dstI = edge + NEDGE;

    // weight repacks (tiny, once per launch)
    repack_w_kernel<<<cdiv(EMB * HDIM, 256), 256, 0, stream>>>(W0a, w0aR, EMB, HDIM);
    repack_w_kernel<<<cdiv(HDIM * HDIM, 256), 256, 0, stream>>>(W0b, w0bR, HDIM, HDIM);
    repack_w_kernel<<<cdiv(HDIM * HDIM, 256), 256, 0, stream>>>(Wha, whaR, HDIM, HDIM);
    repack_w_kernel<<<cdiv(HDIM * HDIM, 256), 256, 0, stream>>>(Whb, whbR, HDIM, HDIM);

    // embedding gather -> h (N x 64 f32)
    embed_kernel<<<cdiv((long long)NNODE * 16, 256), 256, 0, stream>>>(x, emb, h);

    for (int l = 0; l < 3; ++l) {
        const int Hin = (l == 0) ? EMB : HDIM;
        const int shift = (Hin == EMB) ? 4 : 6;       // log2(Hin/4)
        const unsigned stot = (unsigned)NEDGE << shift;
        const int ztot = NNODE * Hin;

        init_z_kernel<<<cdiv(ztot, 256), 256, 0, stream>>>(h, eps, l, z, ztot);
        scatter_kernel<<<cdiv(stot, 256), 256, 0, stream>>>(h, srcI, dstI, z,
                                                            Hin, shift, stot);
        f32_to_bf16_kernel<<<cdiv(ztot, 256), 256, 0, stream>>>(z, abf, ztot);

        const __bf16* Wa = (l == 0) ? w0aR : whaR;
        const __bf16* Wb = (l == 0) ? w0bR : whbR;
        const float *ba = (l == 0) ? b0a : bha, *gg = (l == 0) ? g0 : gh;
        const float *bb = (l == 0) ? be0 : beh, *b2 = (l == 0) ? b0b : bhb;

        gemm_ln_relu_kernel<<<NNODE / 16, 256, 0, stream>>>(abf, Wa, ba, gg, bb,
                                                            tbf, Hin);
        gemm_bias_kernel<<<NNODE / 16, 256, 0, stream>>>(tbf, Wb, b2, h);
    }

    graph_sum_kernel<<<cdiv(BATCH * 2 * HDIM, 256), 256, 0, stream>>>(h, (float*)d_out);
}
